// TransformerHead_18691697672508
// MI455X (gfx1250) — matrix operand order repaired and checked by reading, not compiled, not run on hardware
//
#include <hip/hip_runtime.h>
#include <hip/hip_bf16.h>

// ---------------------------------------------------------------------------
// Problem dims (match reference): B=4, S=4096, D=DK=1024
// ---------------------------------------------------------------------------
#define B_   4
#define S_   4096
#define D_   1024
#define DK_  1024
#define M_   (B_ * S_)          // 16384 rows for the QKV projections

// attention tiling
#define BQ   16                 // queries per workgroup (one 16-row WMMA tile)
#define BN   64                 // keys per iteration (4 x 16-col score frags)
#define NW   8                  // waves per workgroup (256 threads, wave32)
#define DSL  128                // head-dim slice owned by each wave (8*128=1024)

typedef __attribute__((ext_vector_type(16))) __bf16         bf16x16;
typedef __attribute__((ext_vector_type(8)))  float          f32x8;
typedef __attribute__((ext_vector_type(8)))  unsigned short u16x8;
typedef __attribute__((ext_vector_type(4)))  unsigned short u16x4;

// 32-byte WMMA operand fragment (16 bf16 per lane), assembled from two
// 16-byte halves (both A and B layouts).
union Frag {
    struct { u16x8 lo, hi; } h;
    bf16x16 v;
};

__device__ __forceinline__ unsigned short f32_to_bf16(float f) {
    unsigned int u = __float_as_uint(f);
    u += 0x7FFFu + ((u >> 16) & 1u);        // round-to-nearest-even
    return (unsigned short)(u >> 16);
}

// A-matrix 16x32 bf16 (ISA 7.12.2): lane L holds row M=L%16, half=L/16;
// VGPR0-3 <- K = half*8 .. +7, VGPR4-7 <- K = 16+half*8 .. +7.
__device__ __forceinline__ Frag load_frag_a(const unsigned short* __restrict__ base,
                                            int ld, int row0, int k0,
                                            int nl, int hf) {
    Frag f;
    const unsigned short* p = base + (size_t)(row0 + nl) * ld + k0;
    f.h.lo = *(const u16x8*)(p + hf * 8);
    f.h.hi = *(const u16x8*)(p + 16 + hf * 8);
    return f;
}

// B-matrix 32x16 bf16: lane L holds column N=L%16, half=L/16;
// VGPR0-3 <- K = half*8 .. +7, VGPR4-7 <- K = 16+half*8 .. +7.
// `base` is the [N][K] row-major source (W rows / K rows / V^T rows).
__device__ __forceinline__ Frag load_frag_b(const unsigned short* __restrict__ base,
                                            int ld, int col0, int k0,
                                            int nl, int hf) {
    Frag f;
    const unsigned short* p = base + (size_t)(col0 + nl) * ld + k0;
    f.h.lo = *(const u16x8*)(p + hf * 8);
    f.h.hi = *(const u16x8*)(p + 16 + hf * 8);
    return f;
}

__device__ __forceinline__ f32x8 wmma_bf16(const Frag& a, const Frag& b, f32x8 c) {
    return __builtin_amdgcn_wmma_f32_16x16x32_bf16(
        /*neg_a=*/false, a.v, /*neg_b=*/false, b.v,
        /*c_mod=*/(short)0, c, /*reuse_a=*/false, /*reuse_b=*/false);
}

// ---------------------------------------------------------------------------
// Kernel 0: f32 -> bf16 cast (x and the three weight matrices)
// ---------------------------------------------------------------------------
__global__ __launch_bounds__(256) void cast_bf16_kernel(
    const float* __restrict__ in, unsigned short* __restrict__ out, int n) {
    int i = (blockIdx.x * 256 + threadIdx.x) * 4;
    if (i + 3 < n) {
        float4 f = *(const float4*)(in + i);
        u16x4 o4 = { f32_to_bf16(f.x), f32_to_bf16(f.y),
                     f32_to_bf16(f.z), f32_to_bf16(f.w) };
        *(u16x4*)(out + i) = o4;
    }
}

// ---------------------------------------------------------------------------
// Kernel 1: bf16 GEMM  Y = X @ W^T   (X:[M_,1024], W:[1024,1024] row-major)
// 8 waves/WG; wave w computes rows [bx*128+w*16, +16) x cols [by*64, +64).
// transposed==1 stores V^T per batch: Vt[b][d][s]  (so attention P@V reads
// contiguous along the key dim).
// ---------------------------------------------------------------------------
__global__ __launch_bounds__(256) void qkv_gemm_kernel(
    const unsigned short* __restrict__ Xb,
    const unsigned short* __restrict__ Wb,
    unsigned short* __restrict__ Yb,
    int transposed) {
    const int lane = threadIdx.x & 31;
    const int wv   = threadIdx.x >> 5;
    const int nl   = lane & 15;
    const int hf   = lane >> 4;
    const int row0 = blockIdx.x * 128 + wv * 16;
    const int col0 = blockIdx.y * 64;

    f32x8 acc[4];
    #pragma unroll
    for (int j = 0; j < 4; ++j) acc[j] = (f32x8){};

    for (int k0 = 0; k0 < D_; k0 += 32) {
        Frag a = load_frag_a(Xb, D_, row0, k0, nl, hf);
        #pragma unroll
        for (int j = 0; j < 4; ++j) {
            Frag b = load_frag_b(Wb, D_, col0 + j * 16, k0, nl, hf);
            acc[j] = wmma_bf16(a, b, acc[j]);
        }
    }

    if (!transposed) {
        #pragma unroll
        for (int j = 0; j < 4; ++j)
            #pragma unroll
            for (int r = 0; r < 8; ++r)
                Yb[(size_t)(row0 + r + 8 * hf) * D_ + col0 + j * 16 + nl] =
                    f32_to_bf16(acc[j][r]);
    } else {
        const int bb = row0 >> 12;          // 128-row tiles never straddle a batch
        const int s0 = row0 & (S_ - 1);
        #pragma unroll
        for (int j = 0; j < 4; ++j)
            #pragma unroll
            for (int r = 0; r < 8; ++r)
                Yb[((size_t)bb * D_ + (col0 + j * 16 + nl)) * S_ + s0 + r + 8 * hf] =
                    f32_to_bf16(acc[j][r]);
    }
}

// ---------------------------------------------------------------------------
// Kernel 2: causal flash attention.
// Grid: (S_/BQ, B_), 256 threads (8 waves).
// Each WG owns a 16-query tile. Wave w owns head-dim slice [w*128, +128) of
// the O accumulator. Scores contract over the full 1024-d, computed as 8
// per-wave partials reduced via ds_add_f32 in LDS. K/V stream from global
// (whole K+V working set = 64 MB, resident in the 192 MB L2).
// ---------------------------------------------------------------------------
__global__ __launch_bounds__(256) void flash_attn_kernel(
    const unsigned short* __restrict__ Q,   // [B][S][1024] bf16
    const unsigned short* __restrict__ K,   // [B][S][1024] bf16
    const unsigned short* __restrict__ Vt,  // [B][1024][S] bf16 (transposed V)
    float* __restrict__ out) {              // [B][S][1024] f32
    __shared__ float          s_red[BQ * BN];           // 4 KB score-reduce tile
    __shared__ unsigned short s_P[NW * BQ * BN];        // 16 KB wave-private P

    const int lane = threadIdx.x & 31;
    const int wv   = threadIdx.x >> 5;
    const int nl   = lane & 15;
    const int hf   = lane >> 4;
    const int q0   = blockIdx.x * BQ;
    const int bb   = blockIdx.y;
    const int dsl  = wv * DSL;

    const unsigned short* Qb = Q  + (size_t)bb * S_ * DK_;
    const unsigned short* Kb = K  + (size_t)bb * S_ * DK_;
    const unsigned short* Vb = Vt + (size_t)bb * D_ * S_;
    float*                Ob = out + (size_t)bb * S_ * D_;

    // Q A-fragments for this wave's 128-d slice, resident for the whole kernel.
    Frag qf[4];
    #pragma unroll
    for (int c = 0; c < 4; ++c)
        qf[c] = load_frag_a(Qb, DK_, q0, dsl + c * 32, nl, hf);

    f32x8 o[8];
    #pragma unroll
    for (int f = 0; f < 8; ++f) o[f] = (f32x8){};
    float m_i[8], l_i[8];
    #pragma unroll
    for (int r = 0; r < 8; ++r) { m_i[r] = -__builtin_inff(); l_i[r] = 0.f; }

    const float inv_sqrt_dk = 0.03125f;     // 1/sqrt(1024)
    const int nblk = (q0 + BQ + BN - 1) / BN;

    for (int blk = 0; blk < nblk; ++blk) {
        const int k0 = blk * BN;

        if (blk + 1 < nblk)                  // global_prefetch_b8 next K block
            __builtin_prefetch(Kb + (size_t)(k0 + BN + nl) * DK_ + dsl, 0, 3);

        // ---- partial scores over our 128-d slice: 16 WMMAs ----
        f32x8 sfr[4];
        #pragma unroll
        for (int nf = 0; nf < 4; ++nf) sfr[nf] = (f32x8){};
        #pragma unroll
        for (int c = 0; c < 4; ++c) {
            #pragma unroll
            for (int nf = 0; nf < 4; ++nf) {
                Frag b = load_frag_b(Kb, DK_, k0 + nf * 16, dsl + c * 32, nl, hf);
                sfr[nf] = wmma_bf16(qf[c], b, sfr[nf]);
            }
        }

        // ---- cross-wave reduction of the 16x64 score tile in LDS ----
        __syncthreads();                                    // s_red reusable
        {
            int t = threadIdx.x * 4;
            s_red[t] = 0.f; s_red[t + 1] = 0.f; s_red[t + 2] = 0.f; s_red[t + 3] = 0.f;
        }
        __syncthreads();
        #pragma unroll
        for (int nf = 0; nf < 4; ++nf)
            #pragma unroll
            for (int r = 0; r < 8; ++r)
                atomicAdd(&s_red[(r + 8 * hf) * BN + nf * 16 + nl], sfr[nf][r]);
        __syncthreads();

        // ---- read full scores, scale, causal mask ----
        const bool diag = (k0 + BN - 1) > q0;
        float sv[4][8];
        #pragma unroll
        for (int nf = 0; nf < 4; ++nf)
            #pragma unroll
            for (int r = 0; r < 8; ++r) {
                float v = s_red[(r + 8 * hf) * BN + nf * 16 + nl] * inv_sqrt_dk;
                if (diag) {
                    int kk = k0 + nf * 16 + nl;
                    int qq = q0 + r + 8 * hf;
                    if (kk > qq) v = -__builtin_inff();
                }
                sv[nf][r] = v;
            }

        // ---- online softmax (rows live in 16-lane halves; xor-shuffle reduce) ----
        float al[8];
        #pragma unroll
        for (int r = 0; r < 8; ++r) {
            float rm = sv[0][r];
            rm = fmaxf(rm, sv[1][r]); rm = fmaxf(rm, sv[2][r]); rm = fmaxf(rm, sv[3][r]);
            #pragma unroll
            for (int off = 8; off >= 1; off >>= 1)
                rm = fmaxf(rm, __shfl_xor(rm, off, 32));
            float nm = fmaxf(m_i[r], rm);
            al[r] = __expf(m_i[r] - nm);
            float rs = 0.f;
            #pragma unroll
            for (int nf = 0; nf < 4; ++nf) {
                float p = __expf(sv[nf][r] - nm);
                rs += p;
                s_P[wv * (BQ * BN) + (r + 8 * hf) * BN + nf * 16 + nl] = f32_to_bf16(p);
            }
            #pragma unroll
            for (int off = 8; off >= 1; off >>= 1)
                rs += __shfl_xor(rs, off, 32);
            l_i[r] = l_i[r] * al[r] + rs;
            m_i[r] = nm;
        }
        #pragma unroll
        for (int f = 0; f < 8; ++f)
            #pragma unroll
            for (int r = 0; r < 8; ++r)
                o[f][r] *= al[r];

        // wave-private LDS round trip: C/D layout -> A-fragment layout
        asm volatile("s_wait_dscnt 0" ::: "memory");
        const unsigned short* pp = &s_P[wv * (BQ * BN) + nl * BN];
        Frag p0, p1;
        p0.h.lo = *(const u16x8*)(pp + hf * 8);
        p0.h.hi = *(const u16x8*)(pp + 16 + hf * 8);
        p1.h.lo = *(const u16x8*)(pp + 32 + hf * 8);
        p1.h.hi = *(const u16x8*)(pp + 48 + hf * 8);

        // ---- O += P @ V  over this wave's 128 channels: 16 WMMAs ----
        #pragma unroll
        for (int f = 0; f < 8; ++f) {
            Frag b0 = load_frag_b(Vb, S_, dsl + f * 16, k0,      nl, hf);
            o[f] = wmma_bf16(p0, b0, o[f]);
            Frag b1 = load_frag_b(Vb, S_, dsl + f * 16, k0 + 32, nl, hf);
            o[f] = wmma_bf16(p1, b1, o[f]);
        }
    }

    // ---- normalize and store ----
    float rl[8];
    #pragma unroll
    for (int r = 0; r < 8; ++r) rl[r] = 1.0f / l_i[r];
    #pragma unroll
    for (int f = 0; f < 8; ++f)
        #pragma unroll
        for (int r = 0; r < 8; ++r)
            Ob[(size_t)(q0 + r + 8 * hf) * D_ + dsl + f * 16 + nl] = o[f][r] * rl[r];
}

// ---------------------------------------------------------------------------
// Host launcher.  Workspace layout (bf16 = 2 B each):
//   xb[16384x1024] | wqb[1024x1024] | wkb | wvb | Qb[16384x1024] | Kb | Vt
//   total ~140.5 MB.
// ---------------------------------------------------------------------------
extern "C" void kernel_launch(void* const* d_in, const int* in_sizes, int n_in,
                              void* d_out, int out_size, void* d_ws, size_t ws_size,
                              hipStream_t stream) {
    const float* x  = (const float*)d_in[0];
    const float* Wq = (const float*)d_in[1];
    const float* Wk = (const float*)d_in[2];
    const float* Wv = (const float*)d_in[3];
    float* out = (float*)d_out;

    unsigned short* xb  = (unsigned short*)d_ws;
    unsigned short* wqb = xb  + (size_t)M_ * D_;
    unsigned short* wkb = wqb + (size_t)D_ * D_;
    unsigned short* wvb = wkb + (size_t)D_ * D_;
    unsigned short* Qb  = wvb + (size_t)D_ * D_;
    unsigned short* Kb  = Qb  + (size_t)M_ * DK_;
    unsigned short* Vt  = Kb  + (size_t)M_ * DK_;

    cast_bf16_kernel<<<(M_ * D_) / 1024, 256, 0, stream>>>(x,  xb,  M_ * D_);
    cast_bf16_kernel<<<(D_ * D_) / 1024, 256, 0, stream>>>(Wq, wqb, D_ * D_);
    cast_bf16_kernel<<<(D_ * D_) / 1024, 256, 0, stream>>>(Wk, wkb, D_ * D_);
    cast_bf16_kernel<<<(D_ * D_) / 1024, 256, 0, stream>>>(Wv, wvb, D_ * D_);

    dim3 gg(M_ / 128, D_ / 64);
    qkv_gemm_kernel<<<gg, 256, 0, stream>>>(xb, wqb, Qb, 0);
    qkv_gemm_kernel<<<gg, 256, 0, stream>>>(xb, wkb, Kb, 0);
    qkv_gemm_kernel<<<gg, 256, 0, stream>>>(xb, wvb, Vt, 1);   // stores V^T

    dim3 ga(S_ / BQ, B_);
    flash_attn_kernel<<<ga, 256, 0, stream>>>(Qb, Kb, Vt, out);
}
